// EquivariantGNN_14001593385582
// MI455X (gfx1250) — matrix operand order repaired and checked by reading, not compiled, not run on hardware
//
#include <hip/hip_runtime.h>
#include <math.h>

typedef __attribute__((ext_vector_type(16))) __bf16 v16bf;
typedef __attribute__((ext_vector_type(8)))  __bf16 v8bf;
typedef __attribute__((ext_vector_type(8)))  float  v8f;

union AV { v16bf v; v8bf h[2]; __bf16 e[16]; };

__device__ __forceinline__ __bf16 f2bf(float f){
  unsigned u = __builtin_bit_cast(unsigned, f);
  unsigned r = u + 0x7FFFu + ((u >> 16) & 1u);
  unsigned short hs = (unsigned short)(r >> 16);
  return __builtin_bit_cast(__bf16, hs);
}
__device__ __forceinline__ float silu_f(float v){ return v * (1.f/(1.f + __expf(-v))); }
// LDS ops from one wave are processed in order; the asm fence stops the
// compiler from reordering ds reads/writes around the transpose round-trip.
__device__ __forceinline__ void lds_fence(){ asm volatile("s_wait_dscnt 0" ::: "memory"); }

// ---------------------------------------------------------------------------
// Weight prep: f32 [L,K,N] row-major  ->  bf16 [L,Npad,Kpad]  (n-major, padded)
// ---------------------------------------------------------------------------
__global__ void wprep_kernel(const float* __restrict__ src, __bf16* __restrict__ dst,
                             int Lw, int K, int Nn, int Kpad, int Npad){
  long total = (long)Lw * Npad * Kpad;
  long stride = (long)gridDim.x * blockDim.x;
  for (long i = (long)blockIdx.x*blockDim.x + threadIdx.x; i < total; i += stride){
    int kp = (int)(i % Kpad);
    long t = i / Kpad;
    int n  = (int)(t % Npad);
    int l  = (int)(t / Npad);
    float v = 0.f;
    if (kp < K && n < Nn) v = src[((long)l*K + kp)*Nn + n];
    dst[i] = f2bf(v);
  }
}

__global__ void prep_layer_kernel(const float* __restrict__ hin, __bf16* __restrict__ hbf,
                                  float* __restrict__ aggH, float* __restrict__ aggX,
                                  float* __restrict__ gsum, int nd, int n3){
  int stride = gridDim.x * blockDim.x;
  for (int i = blockIdx.x*blockDim.x + threadIdx.x; i < nd; i += stride){
    hbf[i] = f2bf(hin[i]);
    aggH[i] = 0.f;
  }
  for (int i = blockIdx.x*blockDim.x + threadIdx.x; i < n3; i += stride) aggX[i] = 0.f;
  if (blockIdx.x == 0 && threadIdx.x == 0) gsum[0] = 0.f;
}

__global__ void bf_convert_kernel(const float* __restrict__ src, __bf16* __restrict__ dst, int n){
  int stride = gridDim.x * blockDim.x;
  for (int i = blockIdx.x*blockDim.x + threadIdx.x; i < n; i += stride) dst[i] = f2bf(src[i]);
}

__global__ void gnorm_kernel(const int* __restrict__ eSrc, const int* __restrict__ eDst,
                             const float* __restrict__ xin, float* __restrict__ gsum, int ne){
  float acc = 0.f;
  int stride = gridDim.x * blockDim.x;
  for (int i = blockIdx.x*blockDim.x + threadIdx.x; i < ne; i += stride){
    int s = eSrc[i], d = eDst[i];
    float dx = xin[s*3+0] - xin[d*3+0];
    float dy = xin[s*3+1] - xin[d*3+1];
    float dz = xin[s*3+2] - xin[d*3+2];
    acc += dx*dx + dy*dy + dz*dz;
  }
  #pragma unroll
  for (int m = 16; m >= 1; m >>= 1) acc += __shfl_xor(acc, m, 32);
  if ((threadIdx.x & 31) == 0) atomicAdd(gsum, acc);
}

// ---------------------------------------------------------------------------
// Edge kernel: one wave = 16 edges.  feat(16x288 bf16) kept in regs (A-layout).
//   m  = silu(feat@Wm1+bm1)@Wm2+bm2 ; m *= sigmoid(m.Wa+ba) ; atomic-> aggH
//   mx = silu(feat@Wx1+bx1)@Wx2+bx2 ; atomic diff*mx -> aggX (scale deferred)
// ---------------------------------------------------------------------------
__global__ __launch_bounds__(256) void edge_kernel(
    const int* __restrict__ eSrc, const int* __restrict__ eDst,
    const float* __restrict__ xin, const __bf16* __restrict__ hbf,
    const __bf16* __restrict__ wm1, const float* __restrict__ bm1,
    const __bf16* __restrict__ wm2, const float* __restrict__ bm2,
    const __bf16* __restrict__ wx1, const float* __restrict__ bx1,
    const __bf16* __restrict__ wx2, const float* __restrict__ bx2,
    const float* __restrict__ wa, const float* __restrict__ bap,
    float* __restrict__ aggH, float* __restrict__ aggX)
{
  __shared__ __align__(32) __bf16 sHid[8][16*256];   // 8 KB per wave
  const int lane = threadIdx.x & 31;
  const int w    = threadIdx.x >> 5;
  const int tile = blockIdx.x * 8 + w;
  const int eb   = tile * 16;
  const int half = (lane >> 4) & 1;
  const int n0   = lane & 15;
  const int row  = lane & 15;
  const int kb   = half ? 8  : 0;   // A-layout k base
  const int kbB  = half ? 8  : 0;   // B-layout k base
  const int moff = half ? 8  : 0;   // C-layout row offset
  __bf16* hid = &sHid[w][0];

  // per-lane edge endpoints (lanes 0..15 own edge `row`; 16..31 mirror)
  const int mySrc = eSrc[eb + row];
  const int myDst = eDst[eb + row];
  float dx = xin[mySrc*3+0] - xin[myDst*3+0];
  float dy = xin[mySrc*3+1] - xin[myDst*3+1];
  float dz = xin[mySrc*3+2] - xin[myDst*3+2];
  float d2 = dx*dx + dy*dy + dz*dz;

  // ---- A = feat tile, 9 k-steps, loaded straight into WMMA A-layout
  AV af[9];
  {
    const __bf16* pS = hbf + (long)mySrc * 128;
    const __bf16* pD = hbf + (long)myDst * 128;
    #pragma unroll
    for (int s = 0; s < 4; ++s){
      af[s].h[0] = *(const v8bf*)(pS + 32*s + kb);
      af[s].h[1] = *(const v8bf*)(pS + 32*s + 16 + kb);
    }
    #pragma unroll
    for (int s = 4; s < 8; ++s){
      af[s].h[0] = *(const v8bf*)(pD + 32*(s-4) + kb);
      af[s].h[1] = *(const v8bf*)(pD + 32*(s-4) + 16 + kb);
    }
    const __bf16 z = __builtin_bit_cast(__bf16, (unsigned short)0);
    #pragma unroll
    for (int j = 0; j < 16; ++j) af[8].e[j] = z;
    if (!half) af[8].e[0] = f2bf(d2);   // k==256 column
  }

  // first-layer GEMM: feat(16x288) @ W(288x256) -> silu -> LDS (bf16)
  auto gemm1 = [&](const __bf16* W, const float* bias){
    #pragma unroll 1
    for (int nt = 0; nt < 16; ++nt){
      const int n = nt*16 + n0;
      const float bv = bias[n];
      v8f c;
      #pragma unroll
      for (int r = 0; r < 8; ++r) c[r] = bv;
      const __bf16* wb = W + (long)n * 288 + kbB;
      #pragma unroll
      for (int s = 0; s < 9; ++s){
        v16bf b = __builtin_shufflevector(*(const v8bf*)(wb + 32*s), *(const v8bf*)(wb + 32*s + 16), 0,1,2,3,4,5,6,7,8,9,10,11,12,13,14,15);
        c = __builtin_amdgcn_wmma_f32_16x16x32_bf16(false, af[s].v, false, b,
                                                    (short)0, c, false, false);
      }
      #pragma unroll
      for (int r = 0; r < 8; ++r)
        hid[(r + moff)*256 + n] = f2bf(silu_f(c[r]));
    }
  };

  const __bf16* hrow = hid + row*256;

  // ========================= x-message chain =========================
  gemm1(wx1, bx1);
  lds_fence();
  {
    AV ax[8];
    #pragma unroll
    for (int s = 0; s < 8; ++s){
      ax[s].h[0] = *(const v8bf*)(hrow + 32*s + kb);
      ax[s].h[1] = *(const v8bf*)(hrow + 32*s + 16 + kb);
    }
    v8f c;
    #pragma unroll
    for (int r = 0; r < 8; ++r) c[r] = (n0 < 3) ? bx2[n0] : 0.f;
    const __bf16* wb = wx2 + (long)n0 * 256 + kbB;
    #pragma unroll
    for (int s = 0; s < 8; ++s){
      v16bf b = __builtin_shufflevector(*(const v8bf*)(wb + 32*s), *(const v8bf*)(wb + 32*s + 16), 0,1,2,3,4,5,6,7,8,9,10,11,12,13,14,15);
      c = __builtin_amdgcn_wmma_f32_16x16x32_bf16(false, ax[s].v, false, b,
                                                  (short)0, c, false, false);
    }
    // xm = diff * mx (division by (gnorm+1) deferred to node kernel)
    #pragma unroll
    for (int r = 0; r < 8; ++r){
      float cr = c[r];
      float q0 = __shfl(cr, 0, 32),  q1 = __shfl(cr, 1, 32),  q2 = __shfl(cr, 2, 32);
      float p0 = __shfl(cr, 16, 32), p1 = __shfl(cr, 17, 32), p2 = __shfl(cr, 18, 32);
      if (lane == r){
        atomicAdd(&aggX[(long)mySrc*3+0], dx*q0);
        atomicAdd(&aggX[(long)mySrc*3+1], dy*q1);
        atomicAdd(&aggX[(long)mySrc*3+2], dz*q2);
      } else if (lane == r + 8){
        atomicAdd(&aggX[(long)mySrc*3+0], dx*p0);
        atomicAdd(&aggX[(long)mySrc*3+1], dy*p1);
        atomicAdd(&aggX[(long)mySrc*3+2], dz*p2);
      }
    }
  }

  // ========================= h-message chain =========================
  lds_fence();            // WAR on hid
  gemm1(wm1, bm1);
  lds_fence();
  AV am[8];
  #pragma unroll
  for (int s = 0; s < 8; ++s){
    am[s].h[0] = *(const v8bf*)(hrow + 32*s + kb);
    am[s].h[1] = *(const v8bf*)(hrow + 32*s + 16 + kb);
  }
  v8f mc[8];
  #pragma unroll
  for (int nt = 0; nt < 8; ++nt){
    const int n = nt*16 + n0;
    v8f c;
    #pragma unroll
    for (int r = 0; r < 8; ++r) c[r] = bm2[n];
    const __bf16* wb = wm2 + (long)n * 256 + kbB;
    #pragma unroll
    for (int s = 0; s < 8; ++s){
      v16bf b = __builtin_shufflevector(*(const v8bf*)(wb + 32*s), *(const v8bf*)(wb + 32*s + 16), 0,1,2,3,4,5,6,7,8,9,10,11,12,13,14,15);
      c = __builtin_amdgcn_wmma_f32_16x16x32_bf16(false, am[s].v, false, b,
                                                  (short)0, c, false, false);
    }
    mc[nt] = c;
  }
  // attention gate: p[M] = sum_N m[M,N]*wa[N]; reduce across the 16-lane half
  float p[8];
  #pragma unroll
  for (int r = 0; r < 8; ++r) p[r] = 0.f;
  #pragma unroll
  for (int nt = 0; nt < 8; ++nt){
    float wv = wa[nt*16 + n0];
    #pragma unroll
    for (int r = 0; r < 8; ++r) p[r] += mc[nt][r] * wv;
  }
  #pragma unroll
  for (int msk = 1; msk <= 8; msk <<= 1){
    #pragma unroll
    for (int r = 0; r < 8; ++r) p[r] += __shfl_xor(p[r], msk, 32);
  }
  const float ba0 = bap[0];
  float g[8];
  #pragma unroll
  for (int r = 0; r < 8; ++r) g[r] = 1.f/(1.f + __expf(-(p[r] + ba0)));
  // scatter gated messages into aggH
  #pragma unroll
  for (int r = 0; r < 8; ++r){
    const int node = eSrc[eb + r + moff];
    float* dstp = aggH + (long)node*128 + n0;
    #pragma unroll
    for (int nt = 0; nt < 8; ++nt)
      atomicAdd(dstp + nt*16, mc[nt][r] * g[r]);
  }
}

// ---------------------------------------------------------------------------
// Node kernel: one wave = 16 nodes.
//   new_h = silu([h|aggH]@Wh1+bh1)@Wh2+bh2 ; new_x = x + aggX/(sqrt(gsum)+1)
// ---------------------------------------------------------------------------
__global__ __launch_bounds__(256) void node_kernel(
    const __bf16* __restrict__ hbf, const __bf16* __restrict__ aggbf,
    const __bf16* __restrict__ wh1, const float* __restrict__ bh1,
    const __bf16* __restrict__ wh2, const float* __restrict__ bh2,
    const float* __restrict__ xin, const float* __restrict__ aggX,
    const float* __restrict__ gsum,
    float* __restrict__ hout, float* __restrict__ xout, int nTiles)
{
  __shared__ __align__(32) __bf16 sHid[8][16*256];
  const int lane = threadIdx.x & 31;
  const int w    = threadIdx.x >> 5;
  const int tile = blockIdx.x * 8 + w;
  if (tile >= nTiles) return;          // whole-wave exit; no block barriers used
  const int nb   = tile * 16;
  const int half = (lane >> 4) & 1;
  const int n0   = lane & 15;
  const int row  = lane & 15;
  const int kb   = half ? 8  : 0;
  const int kbB  = half ? 8  : 0;
  const int moff = half ? 8  : 0;
  __bf16* hid = &sHid[w][0];

  const int node = nb + row;
  AV a[8];
  {
    const __bf16* pH = hbf   + (long)node * 128;
    const __bf16* pA = aggbf + (long)node * 128;
    #pragma unroll
    for (int s = 0; s < 4; ++s){
      a[s].h[0] = *(const v8bf*)(pH + 32*s + kb);
      a[s].h[1] = *(const v8bf*)(pH + 32*s + 16 + kb);
    }
    #pragma unroll
    for (int s = 4; s < 8; ++s){
      a[s].h[0] = *(const v8bf*)(pA + 32*(s-4) + kb);
      a[s].h[1] = *(const v8bf*)(pA + 32*(s-4) + 16 + kb);
    }
  }
  // GEMM 1: [h|agg](16x256) @ Wh1(256x256) -> silu -> LDS
  #pragma unroll 1
  for (int nt = 0; nt < 16; ++nt){
    const int n = nt*16 + n0;
    v8f c;
    #pragma unroll
    for (int r = 0; r < 8; ++r) c[r] = bh1[n];
    const __bf16* wb = wh1 + (long)n * 256 + kbB;
    #pragma unroll
    for (int s = 0; s < 8; ++s){
      v16bf b = __builtin_shufflevector(*(const v8bf*)(wb + 32*s), *(const v8bf*)(wb + 32*s + 16), 0,1,2,3,4,5,6,7,8,9,10,11,12,13,14,15);
      c = __builtin_amdgcn_wmma_f32_16x16x32_bf16(false, a[s].v, false, b,
                                                  (short)0, c, false, false);
    }
    #pragma unroll
    for (int r = 0; r < 8; ++r)
      hid[(r + moff)*256 + n] = f2bf(silu_f(c[r]));
  }
  lds_fence();
  AV ah[8];
  const __bf16* hrow = hid + row*256;
  #pragma unroll
  for (int s = 0; s < 8; ++s){
    ah[s].h[0] = *(const v8bf*)(hrow + 32*s + kb);
    ah[s].h[1] = *(const v8bf*)(hrow + 32*s + 16 + kb);
  }
  // GEMM 2: hid(16x256) @ Wh2(256x128) -> new_h
  #pragma unroll 1
  for (int nt = 0; nt < 8; ++nt){
    const int n = nt*16 + n0;
    v8f c;
    #pragma unroll
    for (int r = 0; r < 8; ++r) c[r] = bh2[n];
    const __bf16* wb = wh2 + (long)n * 256 + kbB;
    #pragma unroll
    for (int s = 0; s < 8; ++s){
      v16bf b = __builtin_shufflevector(*(const v8bf*)(wb + 32*s), *(const v8bf*)(wb + 32*s + 16), 0,1,2,3,4,5,6,7,8,9,10,11,12,13,14,15);
      c = __builtin_amdgcn_wmma_f32_16x16x32_bf16(false, ah[s].v, false, b,
                                                  (short)0, c, false, false);
    }
    #pragma unroll
    for (int r = 0; r < 8; ++r)
      hout[(long)(nb + r + moff)*128 + n] = c[r];
  }
  // coordinate update
  if (lane < 16){
    const int nd = nb + lane;
    const float inv = 1.f/(sqrtf(gsum[0]) + 1.f);
    xout[nd*3+0] = xin[nd*3+0] + aggX[nd*3+0]*inv;
    xout[nd*3+1] = xin[nd*3+1] + aggX[nd*3+1]*inv;
    xout[nd*3+2] = xin[nd*3+2] + aggX[nd*3+2]*inv;
  }
}

// ---------------------------------------------------------------------------
extern "C" void kernel_launch(void* const* d_in, const int* in_sizes, int n_in,
                              void* d_out, int out_size, void* d_ws, size_t ws_size,
                              hipStream_t stream)
{
  (void)in_sizes; (void)n_in; (void)out_size; (void)ws_size;
  const int N_ = 20000, E_ = 320000, Dd = 128, Ll = 2;

  const int*   eSrc = (const int*)d_in[0];
  const int*   eDst = eSrc + E_;
  const float* h0   = (const float*)d_in[1];
  const float* x0   = (const float*)d_in[2];
  const float* Wm1  = (const float*)d_in[3];
  const float* bm1  = (const float*)d_in[4];
  const float* Wm2  = (const float*)d_in[5];
  const float* bm2  = (const float*)d_in[6];
  const float* Wx1  = (const float*)d_in[7];
  const float* bx1  = (const float*)d_in[8];
  const float* Wx2  = (const float*)d_in[9];
  const float* bx2  = (const float*)d_in[10];
  const float* Wh1  = (const float*)d_in[11];
  const float* bh1  = (const float*)d_in[12];
  const float* Wh2  = (const float*)d_in[13];
  const float* bh2  = (const float*)d_in[14];
  const float* Wa   = (const float*)d_in[15];
  const float* ba   = (const float*)d_in[16];

  char* base = (char*)d_ws;
  size_t off = 0;
  auto carve = [&](size_t bytes)->char*{
    char* p = base + off; off += (bytes + 255) & ~(size_t)255; return p;
  };
  __bf16* wm1p  = (__bf16*)carve((size_t)Ll*256*288*2);
  __bf16* wx1p  = (__bf16*)carve((size_t)Ll*256*288*2);
  __bf16* wm2p  = (__bf16*)carve((size_t)Ll*128*256*2);
  __bf16* wx2p  = (__bf16*)carve((size_t)Ll*16*256*2);
  __bf16* wh1p  = (__bf16*)carve((size_t)Ll*256*256*2);
  __bf16* wh2p  = (__bf16*)carve((size_t)Ll*128*256*2);
  __bf16* hbf   = (__bf16*)carve((size_t)N_*128*2);
  __bf16* aggbf = (__bf16*)carve((size_t)N_*128*2);
  float*  aggH  = (float*)carve((size_t)N_*128*4);
  float*  aggX  = (float*)carve((size_t)N_*3*4);
  float*  gsum  = (float*)carve(256);
  float*  hA    = (float*)carve((size_t)N_*128*4);
  float*  xA    = (float*)carve((size_t)N_*3*4);

  auto wlaunch = [&](const float* src, __bf16* dst, int K, int Nn, int Kpad, int Npad){
    long total = (long)Ll * Npad * Kpad;
    int blocks = (int)((total + 255)/256);
    wprep_kernel<<<blocks, 256, 0, stream>>>(src, dst, Ll, K, Nn, Kpad, Npad);
  };
  wlaunch(Wm1, wm1p, 257, 256, 288, 256);
  wlaunch(Wx1, wx1p, 257, 256, 288, 256);
  wlaunch(Wm2, wm2p, 256, 128, 256, 128);
  wlaunch(Wx2, wx2p, 256,   3, 256,  16);
  wlaunch(Wh1, wh1p, 256, 256, 256, 256);
  wlaunch(Wh2, wh2p, 256, 128, 256, 128);

  const float* hin  = h0;
  const float* xinp = x0;
  float* outH = (float*)d_out;
  float* outX = outH + (size_t)N_*Dd;

  for (int l = 0; l < Ll; ++l){
    prep_layer_kernel<<<2048, 256, 0, stream>>>(hin, hbf, aggH, aggX, gsum, N_*128, N_*3);
    gnorm_kernel<<<1024, 256, 0, stream>>>(eSrc, eDst, xinp, gsum, E_);
    edge_kernel<<<E_/(16*8), 256, 0, stream>>>(
        eSrc, eDst, xinp, hbf,
        wm1p + (size_t)l*256*288, bm1 + (size_t)l*256,
        wm2p + (size_t)l*128*256, bm2 + (size_t)l*128,
        wx1p + (size_t)l*256*288, bx1 + (size_t)l*256,
        wx2p + (size_t)l*16*256,  bx2 + (size_t)l*3,
        Wa + (size_t)l*128, ba + l,
        aggH, aggX);
    bf_convert_kernel<<<2048, 256, 0, stream>>>(aggH, aggbf, N_*128);
    float* ho = (l == Ll-1) ? outH : hA;
    float* xo = (l == Ll-1) ? outX : xA;
    int nTiles = N_/16;
    node_kernel<<<(nTiles + 7)/8, 256, 0, stream>>>(
        hbf, aggbf,
        wh1p + (size_t)l*256*256, bh1 + (size_t)l*256,
        wh2p + (size_t)l*128*256, bh2 + (size_t)l*128,
        xinp, aggX, gsum, ho, xo, nTiles);
    hin = ho; xinp = xo;
  }
}
